// TransformerBlock_74698071212183
// MI455X (gfx1250) — matrix operand order repaired and checked by reading, not compiled, not run on hardware
//
#include <hip/hip_runtime.h>

// ---------------------------------------------------------------------------
// Transformer block for MI455X (gfx1250, wave32, WMMA).
// All GEMMs run on v_wmma_f32_16x16x32_bf16 (bf16 in, fp32 accumulate).
// Norms / RoPE / softmax / residuals in fp32.
// Attention: single fused flash-attention kernel (online softmax, WMMA for
// both Q*K^T and P*V), 2048 independent waves -- no materialized scores.
// ---------------------------------------------------------------------------

#define S_LEN   2048
#define D_MODEL 1024
#define N_HEAD  16
#define D_HEAD  64
#define D_FFN   4096
#define PSTR    72     // LDS row pitch (u16) for staged P tile: bank-conflict free

typedef unsigned short u16;
typedef __attribute__((ext_vector_type(16))) __bf16 v16bf;
typedef __attribute__((ext_vector_type(8)))  float  v8f;
typedef __attribute__((ext_vector_type(4)))  float  vf4;
typedef __attribute__((ext_vector_type(4)))  u16    vu4;

__device__ __forceinline__ u16 f2bf(float x) {
  // round-to-nearest-even fp32 -> bf16
  unsigned int u = __float_as_uint(x);
  u += 0x7fffu + ((u >> 16) & 1u);
  return (u16)(u >> 16);
}

union Frag {
  v16bf v;
  vf4   p[2];   // two 16-byte loads fill one 8-VGPR fragment
};

// ---------------------------------------------------------------------------
// Generic bf16 WMMA GEMM:  C[M][N] = act(scale * A[M][K] @ Bt[N][K]^T) (*mul)(+res)
//   A  : bf16, row-major [M][lda]
//   Bt : bf16, "row = n, cols = k" (weights pre-transposed) [N][ldb]
//   Cf : optional fp32 output, Cb : optional bf16 output (ldc for both)
// Each wave: 64x64 tile = 4x4 accumulators of 16x16.
// ---------------------------------------------------------------------------
template<int WAVES_M, int WAVES_N>
__global__ __launch_bounds__(WAVES_M * WAVES_N * 32)
void gemm_wmma_bf16(const u16* __restrict__ A, int lda,
                    const u16* __restrict__ Bt, int ldb,
                    float* __restrict__ Cf, u16* __restrict__ Cb, int ldc,
                    int M, int N, int K,
                    const float* __restrict__ resid,
                    const float* __restrict__ mulbuf,
                    float scale, int act /*0 none, 1 silu*/)
{
  const int lane = threadIdx.x & 31;
  const int wave = threadIdx.x >> 5;
  const int wm = wave / WAVES_N;
  const int wn = wave % WAVES_N;
  const int tm = (blockIdx.y * WAVES_M + wm) * 64;
  const int tn = (blockIdx.x * WAVES_N + wn) * 64;
  if (tm >= M || tn >= N) return;   // wave-uniform

  const int hl  = lane >> 4;   // half-wave select
  const int l16 = lane & 15;

  v8f acc[4][4] = {};

  for (int k0 = 0; k0 < K; k0 += 32) {
    Frag a[4], b[4];
    const int ka = k0 + hl * 8;    // A: K pattern {ka..ka+7, ka+16..ka+23}
    const int kb = k0 + hl * 8;    // B: K pattern {kb..kb+7, kb+16..kb+23}
#pragma unroll
    for (int i = 0; i < 4; ++i) {
      const u16* ap = A + (size_t)(tm + i * 16 + l16) * lda + ka;
      a[i].p[0] = *(const vf4*)ap;
      a[i].p[1] = *(const vf4*)(ap + 16);
      const u16* bp = Bt + (size_t)(tn + i * 16 + l16) * ldb + kb;
      b[i].p[0] = *(const vf4*)bp;
      b[i].p[1] = *(const vf4*)(bp + 16);
    }
#pragma unroll
    for (int i = 0; i < 4; ++i)
#pragma unroll
      for (int j = 0; j < 4; ++j)
        acc[i][j] = __builtin_amdgcn_wmma_f32_16x16x32_bf16(
            false, a[i].v, false, b[j].v, (short)0, acc[i][j], false, false);
  }

  // Fused epilogue.  C/D layout: VGPR e -> row e + 8*hl, lane l16 -> col.
#pragma unroll
  for (int i = 0; i < 4; ++i) {
#pragma unroll
    for (int j = 0; j < 4; ++j) {
#pragma unroll
      for (int e = 0; e < 8; ++e) {
        const int row = tm + i * 16 + e + hl * 8;
        const int col = tn + j * 16 + l16;
        const size_t idx = (size_t)row * ldc + col;
        float xv = acc[i][j][e] * scale;
        if (act == 1) xv = xv / (1.0f + __expf(-xv));   // silu
        if (mulbuf)   xv *= mulbuf[idx];
        if (resid)    xv += resid[idx];
        if (Cf) Cf[idx] = xv;
        if (Cb) Cb[idx] = f2bf(xv);
      }
    }
  }
}

// ---------------------------------------------------------------------------
// Fused flash attention (one head, 16 query rows per wave, 4 waves/block).
//   q, k : bf16 [s][h*64+d] (post-RoPE)
//   vT   : bf16 [h*64+d][s]
//   attn : bf16 [s][h*64+d] output (softmax(QK^T/8) V)
// Online softmax: row stats per lane (C-layout row e+8*hl), cross-lane
// reductions via shfl_xor within 16-lane halves. P staged through
// wave-private LDS (no barriers) to convert C-layout -> A-layout.
// ---------------------------------------------------------------------------
__global__ __launch_bounds__(128)
void flash_attn(const u16* __restrict__ q, const u16* __restrict__ k,
                const u16* __restrict__ vT, u16* __restrict__ attn)
{
  __shared__ u16 plds[4 * 16 * PSTR];
  const int lane = threadIdx.x & 31;
  const int wave = threadIdx.x >> 5;
  const int h    = blockIdx.y;
  const int m0   = blockIdx.x * 64 + wave * 16;
  const int hl   = lane >> 4;
  const int l16  = lane & 15;

  // Resident Q fragments: rows m0..m0+15, d = 0..63 (two K-chunks of 32)
  Frag qf[2];
#pragma unroll
  for (int c = 0; c < 2; ++c) {
    const u16* ap = q + (size_t)(m0 + l16) * D_MODEL + h * D_HEAD + c * 32 + hl * 8;
    qf[c].p[0] = *(const vf4*)ap;
    qf[c].p[1] = *(const vf4*)(ap + 16);
  }

  v8f o[4] = {};
  float m_s[8], l_s[8];
#pragma unroll
  for (int e = 0; e < 8; ++e) { m_s[e] = -1e30f; l_s[e] = 0.f; }

  u16* pb = plds + wave * (16 * PSTR);

  for (int j0 = 0; j0 < S_LEN; j0 += 64) {
    // ---- S = Q K^T for 16 x 64 keys (fp32 acc) ----
    v8f sacc[4] = {};
#pragma unroll
    for (int nt = 0; nt < 4; ++nt) {
#pragma unroll
      for (int c = 0; c < 2; ++c) {
        Frag bf;
        const u16* bp = k + (size_t)(j0 + nt * 16 + l16) * D_MODEL
                          + h * D_HEAD + c * 32 + hl * 8;
        bf.p[0] = *(const vf4*)bp;
        bf.p[1] = *(const vf4*)(bp + 16);
        sacc[nt] = __builtin_amdgcn_wmma_f32_16x16x32_bf16(
            false, qf[c].v, false, bf.v, (short)0, sacc[nt], false, false);
      }
    }

    // ---- online softmax: new row max, alpha = exp(m_old - m_new) ----
    float alpha[8];
#pragma unroll
    for (int e = 0; e < 8; ++e) {
      float mx = sacc[0][e];
#pragma unroll
      for (int nt = 1; nt < 4; ++nt) mx = fmaxf(mx, sacc[nt][e]);
      mx *= 0.125f;  // scale > 0 commutes with max
      mx = fmaxf(mx, __shfl_xor(mx, 1));
      mx = fmaxf(mx, __shfl_xor(mx, 2));
      mx = fmaxf(mx, __shfl_xor(mx, 4));
      mx = fmaxf(mx, __shfl_xor(mx, 8));
      const float nm = fmaxf(m_s[e], mx);
      alpha[e] = __expf(m_s[e] - nm);
      m_s[e] = nm;
    }

    // ---- P = exp(S/8 - m), row sums, stage P (bf16) into wave-private LDS ----
#pragma unroll
    for (int e = 0; e < 8; ++e) {
      float rs = 0.f;
#pragma unroll
      for (int nt = 0; nt < 4; ++nt) {
        const float p = __expf(sacc[nt][e] * 0.125f - m_s[e]);
        rs += p;
        pb[(e + 8 * hl) * PSTR + nt * 16 + l16] = f2bf(p);
      }
      rs += __shfl_xor(rs, 1);
      rs += __shfl_xor(rs, 2);
      rs += __shfl_xor(rs, 4);
      rs += __shfl_xor(rs, 8);
      l_s[e] = l_s[e] * alpha[e] + rs;
    }

    // ---- rescale running O by alpha ----
#pragma unroll
    for (int dt = 0; dt < 4; ++dt)
#pragma unroll
      for (int e = 0; e < 8; ++e) o[dt][e] *= alpha[e];

    asm volatile("" ::: "memory");  // keep LDS stores ordered before reloads

    // ---- reload P as A-fragments (same-wave LDS: in-order, no barrier) ----
    Frag pf[2];
#pragma unroll
    for (int c = 0; c < 2; ++c) {
      const u16* ap = pb + l16 * PSTR + c * 32 + hl * 8;
      pf[c].p[0] = *(const vf4*)ap;
      pf[c].p[1] = *(const vf4*)(ap + 16);
    }

    // ---- O += P V ----
#pragma unroll
    for (int dt = 0; dt < 4; ++dt) {
#pragma unroll
      for (int c = 0; c < 2; ++c) {
        Frag vf;
        const u16* vp = vT + (size_t)(h * D_HEAD + dt * 16 + l16) * S_LEN
                           + j0 + c * 32 + hl * 8;
        vf.p[0] = *(const vf4*)vp;
        vf.p[1] = *(const vf4*)(vp + 16);
        o[dt] = __builtin_amdgcn_wmma_f32_16x16x32_bf16(
            false, pf[c].v, false, vf.v, (short)0, o[dt], false, false);
      }
    }
  }

  // ---- epilogue: attn = O / l (bf16) ----
#pragma unroll
  for (int e = 0; e < 8; ++e) {
    const float inv = 1.0f / l_s[e];
    const int row = m0 + e + 8 * hl;
#pragma unroll
    for (int dt = 0; dt < 4; ++dt)
      attn[(size_t)row * D_MODEL + h * D_HEAD + dt * 16 + l16] =
          f2bf(o[dt][e] * inv);
  }
}

// ---------------------------------------------------------------------------
// RMSNorm (fp32 in, bf16 out).  One 256-thread block per row of 1024.
// ---------------------------------------------------------------------------
__global__ void rmsnorm_bf16(const float* __restrict__ x,
                             const float* __restrict__ w,
                             u16* __restrict__ out)
{
  const int row = blockIdx.x;
  const int tid = threadIdx.x;
  const float* xr = x + (size_t)row * D_MODEL;
  vf4 v = ((const vf4*)xr)[tid];
  float ss = v.x * v.x + v.y * v.y + v.z * v.z + v.w * v.w;
  __shared__ float red[256];
  red[tid] = ss;
  __syncthreads();
  for (int s = 128; s > 0; s >>= 1) {
    if (tid < s) red[tid] += red[tid + s];
    __syncthreads();
  }
  const float inv = rsqrtf(red[0] * (1.0f / D_MODEL) + 1e-6f);
  vf4 wv = ((const vf4*)w)[tid];
  vu4 o;
  o.x = f2bf(v.x * inv * wv.x);
  o.y = f2bf(v.y * inv * wv.y);
  o.z = f2bf(v.z * inv * wv.z);
  o.w = f2bf(v.w * inv * wv.w);
  ((vu4*)(out + (size_t)row * D_MODEL))[tid] = o;
}

// ---------------------------------------------------------------------------
// RoPE on fp32 q/k, emitting bf16 (same [s][h][d] layout). 1 thread per pair.
// ---------------------------------------------------------------------------
__global__ void rope_bf16(const float* __restrict__ in,
                          const float* __restrict__ cs,
                          const float* __restrict__ sn,
                          u16* __restrict__ out)
{
  const int idx = blockIdx.x * blockDim.x + threadIdx.x;
  if (idx >= S_LEN * N_HEAD * (D_HEAD / 2)) return;
  const int i = idx & 31;
  const int h = (idx >> 5) & 15;
  const int s = idx >> 9;
  const size_t base = (size_t)s * D_MODEL + h * D_HEAD + i * 2;
  const float te = in[base], to = in[base + 1];
  const float c = cs[s * 32 + i], sv = sn[s * 32 + i];
  const unsigned int pk = (unsigned int)f2bf(te * c - to * sv)
                        | ((unsigned int)f2bf(te * sv + to * c) << 16);
  *(unsigned int*)(out + base) = pk;
}

// ---------------------------------------------------------------------------
// fp32 [K][N] -> bf16 [N][K] tiled transpose (weight prep).
// ---------------------------------------------------------------------------
__global__ void wtrans(const float* __restrict__ in, u16* __restrict__ out,
                       int K, int N)
{
  __shared__ float t[32][33];
  const int n0 = blockIdx.x * 32, k0 = blockIdx.y * 32;
  const int tx = threadIdx.x, ty = threadIdx.y;
#pragma unroll
  for (int r = 0; r < 32; r += 8)
    t[ty + r][tx] = in[(size_t)(k0 + ty + r) * N + n0 + tx];
  __syncthreads();
#pragma unroll
  for (int r = 0; r < 32; r += 8)
    out[(size_t)(n0 + ty + r) * K + k0 + tx] = f2bf(t[tx][ty + r]);
}

// bf16 [R][C] -> bf16 [C][R] tiled transpose (V -> V^T per head layout).
__global__ void btrans(const u16* __restrict__ in, u16* __restrict__ out,
                       int R, int C)
{
  __shared__ u16 t[32][33];
  const int c0 = blockIdx.x * 32, r0 = blockIdx.y * 32;
  const int tx = threadIdx.x, ty = threadIdx.y;
#pragma unroll
  for (int r = 0; r < 32; r += 8)
    t[ty + r][tx] = in[(size_t)(r0 + ty + r) * C + c0 + tx];
  __syncthreads();
#pragma unroll
  for (int r = 0; r < 32; r += 8)
    out[(size_t)(c0 + ty + r) * R + r0 + tx] = t[tx][ty + r];
}

// ---------------------------------------------------------------------------
extern "C" void kernel_launch(void* const* d_in, const int* in_sizes, int n_in,
                              void* d_out, int out_size, void* d_ws, size_t ws_size,
                              hipStream_t stream)
{
  (void)in_sizes; (void)n_in; (void)out_size; (void)ws_size;
  const float* x    = (const float*)d_in[0];
  const float* cosb = (const float*)d_in[1];
  const float* sinb = (const float*)d_in[2];
  const float* wq   = (const float*)d_in[3];
  const float* wk   = (const float*)d_in[4];
  const float* wv   = (const float*)d_in[5];
  const float* wo   = (const float*)d_in[6];
  const float* w1   = (const float*)d_in[7];
  const float* w2   = (const float*)d_in[8];
  const float* w3   = (const float*)d_in[9];
  const float* anw  = (const float*)d_in[10];
  const float* fnw  = (const float*)d_in[11];
  float* out = (float*)d_out;

  char* ws = (char*)d_ws;
  size_t off = 0;
  auto alloc = [&](size_t bytes) -> char* {
    char* p = ws + off;
    off += (bytes + 255) & ~(size_t)255;
    return p;
  };

  // bf16 transposed weights (Wt[n][k])
  u16* wq_t = (u16*)alloc((size_t)D_MODEL * D_MODEL * 2);
  u16* wk_t = (u16*)alloc((size_t)D_MODEL * D_MODEL * 2);
  u16* wv_t = (u16*)alloc((size_t)D_MODEL * D_MODEL * 2);
  u16* wo_t = (u16*)alloc((size_t)D_MODEL * D_MODEL * 2);
  u16* w1_t = (u16*)alloc((size_t)D_FFN * D_MODEL * 2);
  u16* w3_t = (u16*)alloc((size_t)D_FFN * D_MODEL * 2);
  u16* w2_t = (u16*)alloc((size_t)D_MODEL * D_FFN * 2);
  // activations
  u16*   hin  = (u16*)alloc((size_t)S_LEN * D_MODEL * 2);   // rmsnorm(x) bf16
  float* qf   = (float*)alloc((size_t)S_LEN * D_MODEL * 4); // q pre-rope fp32
  float* kf   = (float*)alloc((size_t)S_LEN * D_MODEL * 4); // k pre-rope fp32
  u16*   vb   = (u16*)alloc((size_t)S_LEN * D_MODEL * 2);   // v bf16 [s][h d]
  u16*   qb   = (u16*)alloc((size_t)S_LEN * D_MODEL * 2);   // q post-rope bf16
  u16*   kb   = (u16*)alloc((size_t)S_LEN * D_MODEL * 2);   // k post-rope bf16
  u16*   vT   = (u16*)alloc((size_t)D_MODEL * S_LEN * 2);   // v^T [h d][s]
  u16*   attn = (u16*)alloc((size_t)S_LEN * D_MODEL * 2);   // attention out bf16
  float* hb   = (float*)alloc((size_t)S_LEN * D_MODEL * 4); // h = x + attn@wo
  u16*   fb   = (u16*)alloc((size_t)S_LEN * D_MODEL * 2);   // rmsnorm(h) bf16
  float* t1   = (float*)alloc((size_t)S_LEN * D_FFN * 4);   // silu(f@w1)
  u16*   gb   = (u16*)alloc((size_t)S_LEN * D_FFN * 2);     // t1 * (f@w3) bf16

  const dim3 tb(32, 8);
  // weight prep: fp32 [K][N] -> bf16 [N][K]
  wtrans<<<dim3(D_MODEL/32, D_MODEL/32), tb, 0, stream>>>(wq, wq_t, D_MODEL, D_MODEL);
  wtrans<<<dim3(D_MODEL/32, D_MODEL/32), tb, 0, stream>>>(wk, wk_t, D_MODEL, D_MODEL);
  wtrans<<<dim3(D_MODEL/32, D_MODEL/32), tb, 0, stream>>>(wv, wv_t, D_MODEL, D_MODEL);
  wtrans<<<dim3(D_MODEL/32, D_MODEL/32), tb, 0, stream>>>(wo, wo_t, D_MODEL, D_MODEL);
  wtrans<<<dim3(D_FFN/32,   D_MODEL/32), tb, 0, stream>>>(w1, w1_t, D_MODEL, D_FFN);
  wtrans<<<dim3(D_FFN/32,   D_MODEL/32), tb, 0, stream>>>(w3, w3_t, D_MODEL, D_FFN);
  wtrans<<<dim3(D_MODEL/32, D_FFN/32),   tb, 0, stream>>>(w2, w2_t, D_FFN, D_MODEL);

  // hin = rmsnorm(x) * attn_norm_w
  rmsnorm_bf16<<<S_LEN, 256, 0, stream>>>(x, anw, hin);

  // QKV projections
  gemm_wmma_bf16<2,4><<<dim3(4,16), 256, 0, stream>>>(
      hin, D_MODEL, wq_t, D_MODEL, qf, nullptr, D_MODEL,
      S_LEN, D_MODEL, D_MODEL, nullptr, nullptr, 1.0f, 0);
  gemm_wmma_bf16<2,4><<<dim3(4,16), 256, 0, stream>>>(
      hin, D_MODEL, wk_t, D_MODEL, kf, nullptr, D_MODEL,
      S_LEN, D_MODEL, D_MODEL, nullptr, nullptr, 1.0f, 0);
  gemm_wmma_bf16<2,4><<<dim3(4,16), 256, 0, stream>>>(
      hin, D_MODEL, wv_t, D_MODEL, nullptr, vb, D_MODEL,
      S_LEN, D_MODEL, D_MODEL, nullptr, nullptr, 1.0f, 0);

  // RoPE on q, k (fp32 -> bf16)
  const int nrope = S_LEN * N_HEAD * (D_HEAD / 2);
  rope_bf16<<<(nrope + 255) / 256, 256, 0, stream>>>(qf, cosb, sinb, qb);
  rope_bf16<<<(nrope + 255) / 256, 256, 0, stream>>>(kf, cosb, sinb, kb);

  // v^T for P@V B-fragments: [s][hd] -> [hd][s]
  btrans<<<dim3(D_MODEL/32, S_LEN/32), tb, 0, stream>>>(vb, vT, S_LEN, D_MODEL);

  // Fused flash attention over all heads: grid 32 q-tiles x 16 heads
  flash_attn<<<dim3(S_LEN/64, N_HEAD), 128, 0, stream>>>(qb, kb, vT, attn);

  // h = x + attn @ wo
  gemm_wmma_bf16<2,4><<<dim3(4,16), 256, 0, stream>>>(
      attn, D_MODEL, wo_t, D_MODEL, hb, nullptr, D_MODEL,
      S_LEN, D_MODEL, D_MODEL, x, nullptr, 1.0f, 0);

  // f = rmsnorm(h) * ffn_norm_w
  rmsnorm_bf16<<<S_LEN, 256, 0, stream>>>(hb, fnw, fb);

  // t1 = silu(f @ w1)
  gemm_wmma_bf16<2,4><<<dim3(16,16), 256, 0, stream>>>(
      fb, D_MODEL, w1_t, D_MODEL, t1, nullptr, D_FFN,
      S_LEN, D_FFN, D_MODEL, nullptr, nullptr, 1.0f, 1);
  // g = t1 * (f @ w3)  (bf16)
  gemm_wmma_bf16<2,4><<<dim3(16,16), 256, 0, stream>>>(
      fb, D_MODEL, w3_t, D_MODEL, nullptr, gb, D_FFN,
      S_LEN, D_FFN, D_MODEL, nullptr, t1, 1.0f, 0);
  // out = h + g @ w2
  gemm_wmma_bf16<2,4><<<dim3(4,16), 256, 0, stream>>>(
      gb, D_FFN, w2_t, D_FFN, out, nullptr, D_MODEL,
      S_LEN, D_MODEL, D_FFN, hb, nullptr, 1.0f, 0);
}
